// StagedHead_34823594836247
// MI455X (gfx1250) — compile-verified, matrix operand order checked by reading, not run on hardware
//
#include <hip/hip_runtime.h>
#include <hip/hip_bf16.h>

// ---------------------------------------------------------------------------
// StagedHead two-stage gated cross-attention for MI455X (gfx1250, wave32).
// All matmuls run through v_wmma_f32_16x16x32_bf16 (f32 accumulate).
// GEMMs: double-buffered LDS pipeline (global loads overlap WMMA).
// ---------------------------------------------------------------------------

typedef __attribute__((ext_vector_type(16))) __bf16 bf16x16;
typedef __attribute__((ext_vector_type(8)))  float  f32x8;

#define WMMA_BF16(a, b, c) \
    __builtin_amdgcn_wmma_f32_16x16x32_bf16(false, (a), false, (b), (short)0, (c), false, false)

// Problem constants
#define CB   2
#define CS   2048
#define CT   2048
#define CH   2048
#define CNH  16
#define CNKV 4
#define CHD  128
#define CND  64
#define CKV  512        // NKV*HD
#define CBS  (CB * CS)  // 4096 rows
#define CSCALE 0.08838834764831845f  // 1/sqrt(128)

#define LDS_PITCH 40    // bf16 elems per staged row: 80B, 16B-aligned, padded

// ---------------------------------------------------------------------------
// Fragment loader (CDNA5 16-bit A-matrix layout, 16x32 MxK):
// lane holds row m = lane&15; elements are two contiguous 8-elem runs:
//   k in [h8, h8+8) and [16+h8, 16+h8+8) with h8 = 8*(lane>=16).
// Works for global bf16 rows and LDS-staged rows alike.
// ---------------------------------------------------------------------------
__device__ __forceinline__ bf16x16 ld_row(const __bf16* p, int lane) {
    int h8 = (lane >> 4) << 3;
    bf16x16 f;
#pragma unroll
    for (int i = 0; i < 8; ++i) {
        f[i]     = p[h8 + i];
        f[i + 8] = p[16 + h8 + i];
    }
    return f;
}

// Column fragment from strided bf16 (used for V in attention PV).
__device__ __forceinline__ bf16x16 ld_col(const __bf16* p, int ld, int lane) {
    int h8 = (lane >> 4) << 3;
    bf16x16 f;
#pragma unroll
    for (int i = 0; i < 8; ++i) {
        f[i]     = p[(size_t)(h8 + i) * ld];
        f[i + 8] = p[(size_t)(16 + h8 + i) * ld];
    }
    return f;
}

// ---------------------------------------------------------------------------
// RMSNorm: one block (256 thr = 8 waves) per row; bf16 output.
// ---------------------------------------------------------------------------
__global__ void k_rmsnorm(const float* __restrict__ x, const float* __restrict__ w,
                          __bf16* __restrict__ y) {
    int row = blockIdx.x;
    const float* xr = x + (size_t)row * CH;
    float ss = 0.f;
    for (int i = threadIdx.x; i < CH; i += 256) {
        float v = xr[i];
        ss += v * v;
    }
#pragma unroll
    for (int m = 16; m >= 1; m >>= 1) ss += __shfl_xor(ss, m, 32);
    __shared__ float red[8];
    if ((threadIdx.x & 31) == 0) red[threadIdx.x >> 5] = ss;
    __syncthreads();
    float tot = 0.f;
#pragma unroll
    for (int i = 0; i < 8; ++i) tot += red[i];
    float sc = rsqrtf(tot * (1.0f / CH) + 1e-6f);
    __bf16* yr = y + (size_t)row * CH;
    for (int i = threadIdx.x; i < CH; i += 256)
        yr[i] = (__bf16)(xr[i] * sc * w[i]);
}

// ---------------------------------------------------------------------------
// GEMM: C[M,N] = A[M,K] (AT, row-major) @ W[K,N] (f32, row-major).
// Block = 256 thr (8 waves, 2M x 4N) -> 64x128 C tile; wave = 32x32 (4 accs).
// Double-buffered LDS pipeline:
//   iter: [global-load tile k+1] [WMMA tile k from LDS] [LDS-store tile k+1]
// A staged as [row][k] bf16; W staged transposed [col][k] bf16 so every
// fragment read is two ds_load_b128.
// RESID: outF = resid + sigmoid(gate)*C, else outB = bf16(C).
// M%64==0, N%128==0, K%32==0 at all call sites.
// ---------------------------------------------------------------------------
template <typename AT, bool RESID>
__global__ void k_gemm(const AT* __restrict__ A, const float* __restrict__ W,
                       const float* __restrict__ resid, const float* __restrict__ gate,
                       float* __restrict__ outF, __bf16* __restrict__ outB,
                       int M, int N, int K) {
    __shared__ __bf16 sA[2][64][LDS_PITCH];
    __shared__ __bf16 sW[2][128][LDS_PITCH];

    int tid  = threadIdx.x;
    int lane = tid & 31;
    int wave = tid >> 5;
    int wr = wave >> 2, wc = wave & 3;
    int n  = lane & 15;
    int hv = lane >> 4;
    int row0 = blockIdx.y * 64;
    int col0 = blockIdx.x * 128;

    // cooperative staging geometry
    int arow = tid >> 2;         // 0..63
    int aseg = (tid & 3) * 8;    // 0,8,16,24
    int wk   = tid >> 3;         // 0..31
    int wcg  = (tid & 7) * 16;   // 0..112
    const AT*    agp = A + (size_t)(row0 + arow) * K + aseg;
    const float* wgp = W + (size_t)wk * N + col0 + wcg;

    float ra[8];
    float rw[16];
    auto ldtile = [&](int k0) {
#pragma unroll
        for (int i = 0; i < 8; ++i) ra[i] = (float)agp[k0 + i];
        const float* wp = wgp + (size_t)k0 * N;
#pragma unroll
        for (int i = 0; i < 16; ++i) rw[i] = wp[i];
    };
    auto sttile = [&](int buf) {
#pragma unroll
        for (int i = 0; i < 8; ++i) sA[buf][arow][aseg + i] = (__bf16)ra[i];
#pragma unroll
        for (int i = 0; i < 16; ++i) sW[buf][wcg + i][wk] = (__bf16)rw[i];
    };

    f32x8 acc[2][2] = {};
    auto compute = [&](int buf) {
        bf16x16 a0 = ld_row(&sA[buf][wr * 32 + n][0], lane);
        bf16x16 a1 = ld_row(&sA[buf][wr * 32 + 16 + n][0], lane);
        bf16x16 b0 = ld_row(&sW[buf][wc * 32 + n][0], lane);
        bf16x16 b1 = ld_row(&sW[buf][wc * 32 + 16 + n][0], lane);
        acc[0][0] = WMMA_BF16(a0, b0, acc[0][0]);
        acc[0][1] = WMMA_BF16(a0, b1, acc[0][1]);
        acc[1][0] = WMMA_BF16(a1, b0, acc[1][0]);
        acc[1][1] = WMMA_BF16(a1, b1, acc[1][1]);
    };

    ldtile(0);
    sttile(0);
    __syncthreads();
    int cur = 0;
    for (int k0 = 32; k0 < K; k0 += 32) {
        ldtile(k0);                       // global loads for next tile (async)
        if (k0 + 32 < K) {                // prefetch next-next tile toward L2
            __builtin_prefetch(agp + k0 + 32, 0, 3);
            __builtin_prefetch(wgp + (size_t)(k0 + 32) * N, 0, 3);
        }
        compute(cur);                     // WMMA on current tile from LDS
        sttile(cur ^ 1);                  // commit loads to the other buffer
        __syncthreads();
        cur ^= 1;
    }
    compute(cur);

    float g = 0.f;
    if (RESID) g = 1.f / (1.f + __expf(-gate[0]));
#pragma unroll
    for (int i = 0; i < 2; ++i)
#pragma unroll
        for (int j = 0; j < 2; ++j)
#pragma unroll
            for (int r = 0; r < 8; ++r) {
                int row = row0 + wr * 32 + i * 16 + r + hv * 8;
                int col = col0 + wc * 32 + j * 16 + n;
                size_t o = (size_t)row * N + col;
                if (RESID)
                    outF[o] = resid[o] + g * acc[i][j][r];
                else
                    outB[o] = (__bf16)acc[i][j][r];
            }
}

// ---------------------------------------------------------------------------
// Coarse attention: per wave, 16 queries x all 64 doc keys. Single-pass
// softmax via width-16 shfl reductions; accumulates prob sums for doc_scores.
// Loads batched ahead of each WMMA group. grid = (S/128, B*NH), block = 256.
// ---------------------------------------------------------------------------
__global__ void k_attn_coarse(const __bf16* __restrict__ Q, const __bf16* __restrict__ Kb,
                              const __bf16* __restrict__ Vb, const float* __restrict__ dmask,
                              float* __restrict__ ds_accum, __bf16* __restrict__ Out) {
    int lane = threadIdx.x & 31;
    int wave = threadIdx.x >> 5;
    int b = blockIdx.y >> 4;
    int h = blockIdx.y & 15;
    int kv = h >> 2;
    int q0 = (blockIdx.x * 8 + wave) * 16;
    int n  = lane & 15;
    int hv = lane >> 4;

    __shared__ __bf16 lds_p[8][16][64];

    const __bf16* qp = Q + ((size_t)(b * CS) + q0 + n) * CH + h * CHD;
    bf16x16 qf[4];
#pragma unroll
    for (int kd = 0; kd < 4; ++kd) qf[kd] = ld_row(qp + kd * 32, lane);

    f32x8 sc[4] = {};
#pragma unroll
    for (int kd = 0; kd < 4; ++kd) {
        bf16x16 kf[4];
#pragma unroll
        for (int j = 0; j < 4; ++j)
            kf[j] = ld_row(Kb + ((size_t)(b * CND) + j * 16 + n) * CKV + kv * CHD + kd * 32,
                           lane);
#pragma unroll
        for (int j = 0; j < 4; ++j) sc[j] = WMMA_BF16(qf[kd], kf[j], sc[j]);
    }

    float p[4][8], mx[8], sm[8], bias[4];
#pragma unroll
    for (int j = 0; j < 4; ++j) bias[j] = (1.f - dmask[b * CND + j * 16 + n]) * -1e9f;
#pragma unroll
    for (int r = 0; r < 8; ++r) mx[r] = -3.0e38f;
#pragma unroll
    for (int j = 0; j < 4; ++j)
#pragma unroll
        for (int r = 0; r < 8; ++r) {
            float s = sc[j][r] * CSCALE + bias[j];
            p[j][r] = s;
            mx[r] = fmaxf(mx[r], s);
        }
#pragma unroll
    for (int r = 0; r < 8; ++r) {
#pragma unroll
        for (int m = 1; m < 16; m <<= 1) mx[r] = fmaxf(mx[r], __shfl_xor(mx[r], m, 16));
        sm[r] = 0.f;
    }
#pragma unroll
    for (int j = 0; j < 4; ++j)
#pragma unroll
        for (int r = 0; r < 8; ++r) {
            float e = __expf(p[j][r] - mx[r]);
            p[j][r] = e;
            sm[r] += e;
        }
#pragma unroll
    for (int r = 0; r < 8; ++r) {
        float t = sm[r];
#pragma unroll
        for (int m = 1; m < 16; m <<= 1) t += __shfl_xor(t, m, 16);
        sm[r] = 1.f / t;
    }
    // probs + doc score accumulation + LDS staging (per-wave region; same-wave
    // DS ops are in-order, so no block barrier needed).
#pragma unroll
    for (int j = 0; j < 4; ++j) {
        float dsum = 0.f;
#pragma unroll
        for (int r = 0; r < 8; ++r) {
            float pv = p[j][r] * sm[r];
            dsum += pv;
            lds_p[wave][r + hv * 8][j * 16 + n] = (__bf16)pv;
        }
        atomicAdd(&ds_accum[b * CND + j * 16 + n], dsum);
    }

    f32x8 oacc[8] = {};
#pragma unroll
    for (int kt = 0; kt < 2; ++kt) {
        bf16x16 pf = ld_row(&lds_p[wave][n][kt * 32], lane);
        const __bf16* vbase = Vb + ((size_t)(b * CND) + kt * 32) * CKV + kv * CHD + n;
#pragma unroll
        for (int g = 0; g < 2; ++g) {
            bf16x16 vf[4];
#pragma unroll
            for (int d = 0; d < 4; ++d) vf[d] = ld_col(vbase + (g * 4 + d) * 16, CKV, lane);
#pragma unroll
            for (int d = 0; d < 4; ++d) oacc[g * 4 + d] = WMMA_BF16(pf, vf[d], oacc[g * 4 + d]);
        }
    }
#pragma unroll
    for (int dt = 0; dt < 8; ++dt)
#pragma unroll
        for (int r = 0; r < 8; ++r) {
            int row = q0 + r + hv * 8;
            Out[((size_t)(b * CS) + row) * CH + h * CHD + dt * 16 + n] = (__bf16)oacc[dt][r];
        }
}

// ---------------------------------------------------------------------------
// Fine-bias: bias[b,t] = log(doc_scores[b, map[b,t]] + eps) + (1-mask)*-1e9
// ---------------------------------------------------------------------------
__global__ void k_fine_bias(const float* __restrict__ ds_accum, const int* __restrict__ map,
                            const float* __restrict__ tmask, float* __restrict__ bias) {
    int idx = blockIdx.x * blockDim.x + threadIdx.x;
    if (idx >= CB * CT) return;
    int b = idx / CT;
    float ds = ds_accum[b * CND + map[idx]] * (1.0f / (CNH * CS));
    bias[idx] = __logf(ds + 1e-6f) + (1.f - tmask[idx]) * -1e9f;
}

// ---------------------------------------------------------------------------
// Fine attention: flash-style over T=2048 keys in 64-key blocks, per-key bias.
// grid = (S/128, B*NH), block = 256 (8 waves).
// ---------------------------------------------------------------------------
__global__ void k_attn_fine(const __bf16* __restrict__ Q, const __bf16* __restrict__ Kb,
                            const __bf16* __restrict__ Vb, const float* __restrict__ fbias,
                            __bf16* __restrict__ Out) {
    int lane = threadIdx.x & 31;
    int wave = threadIdx.x >> 5;
    int b = blockIdx.y >> 4;
    int h = blockIdx.y & 15;
    int kv = h >> 2;
    int q0 = (blockIdx.x * 8 + wave) * 16;
    int n  = lane & 15;
    int hv = lane >> 4;

    __shared__ __bf16 lds_p[8][16][64];

    const __bf16* qp = Q + ((size_t)(b * CS) + q0 + n) * CH + h * CHD;
    bf16x16 qf[4];
#pragma unroll
    for (int kd = 0; kd < 4; ++kd) qf[kd] = ld_row(qp + kd * 32, lane);

    float m_i[8], l_i[8];
    f32x8 oacc[8] = {};
#pragma unroll
    for (int r = 0; r < 8; ++r) { m_i[r] = -3.0e38f; l_i[r] = 0.f; }

    for (int nk = 0; nk < CT; nk += 64) {
        f32x8 sc[4] = {};
#pragma unroll
        for (int kd = 0; kd < 4; ++kd) {
            bf16x16 kf[4];
#pragma unroll
            for (int j = 0; j < 4; ++j)
                kf[j] = ld_row(
                    Kb + ((size_t)(b * CT) + nk + j * 16 + n) * CKV + kv * CHD + kd * 32, lane);
#pragma unroll
            for (int j = 0; j < 4; ++j) sc[j] = WMMA_BF16(qf[kd], kf[j], sc[j]);
        }
        float p[4][8], rm[8], rs[8], bias[4];
#pragma unroll
        for (int j = 0; j < 4; ++j) bias[j] = fbias[b * CT + nk + j * 16 + n];
#pragma unroll
        for (int r = 0; r < 8; ++r) rm[r] = -3.0e38f;
#pragma unroll
        for (int j = 0; j < 4; ++j)
#pragma unroll
            for (int r = 0; r < 8; ++r) {
                float s = sc[j][r] * CSCALE + bias[j];
                p[j][r] = s;
                rm[r] = fmaxf(rm[r], s);
            }
#pragma unroll
        for (int r = 0; r < 8; ++r) {
#pragma unroll
            for (int m = 1; m < 16; m <<= 1) rm[r] = fmaxf(rm[r], __shfl_xor(rm[r], m, 16));
            float mn  = fmaxf(m_i[r], rm[r]);
            float sco = __expf(m_i[r] - mn);
            m_i[r] = mn;
            rm[r]  = mn;
            l_i[r] *= sco;
#pragma unroll
            for (int dt = 0; dt < 8; ++dt) oacc[dt][r] *= sco;
            rs[r] = 0.f;
        }
#pragma unroll
        for (int j = 0; j < 4; ++j)
#pragma unroll
            for (int r = 0; r < 8; ++r) {
                float e = __expf(p[j][r] - rm[r]);
                rs[r] += e;
                lds_p[wave][r + hv * 8][j * 16 + n] = (__bf16)e;
            }
#pragma unroll
        for (int r = 0; r < 8; ++r) {
            float t = rs[r];
#pragma unroll
            for (int m = 1; m < 16; m <<= 1) t += __shfl_xor(t, m, 16);
            l_i[r] += t;
        }
#pragma unroll
        for (int kt = 0; kt < 2; ++kt) {
            bf16x16 pf = ld_row(&lds_p[wave][n][kt * 32], lane);
            const __bf16* vbase = Vb + ((size_t)(b * CT) + nk + kt * 32) * CKV + kv * CHD + n;
#pragma unroll
            for (int g = 0; g < 2; ++g) {
                bf16x16 vf[4];
#pragma unroll
                for (int d = 0; d < 4; ++d) vf[d] = ld_col(vbase + (g * 4 + d) * 16, CKV, lane);
#pragma unroll
                for (int d = 0; d < 4; ++d)
                    oacc[g * 4 + d] = WMMA_BF16(pf, vf[d], oacc[g * 4 + d]);
            }
        }
    }
#pragma unroll
    for (int r = 0; r < 8; ++r) l_i[r] = 1.f / l_i[r];
#pragma unroll
    for (int dt = 0; dt < 8; ++dt)
#pragma unroll
        for (int r = 0; r < 8; ++r) {
            int row = q0 + r + hv * 8;
            Out[((size_t)(b * CS) + row) * CH + h * CHD + dt * 16 + n] =
                (__bf16)(oacc[dt][r] * l_i[r]);
        }
}

// ---------------------------------------------------------------------------
// Host launch
// ---------------------------------------------------------------------------
extern "C" void kernel_launch(void* const* d_in, const int* in_sizes, int n_in,
                              void* d_out, int out_size, void* d_ws, size_t ws_size,
                              hipStream_t stream) {
    const float* hidden = (const float*)d_in[0];
    const float* dkey   = (const float*)d_in[1];
    const float* dval   = (const float*)d_in[2];
    const float* dmask  = (const float*)d_in[3];
    const float* tkey   = (const float*)d_in[4];
    const float* tval   = (const float*)d_in[5];
    const float* tmask  = (const float*)d_in[6];
    const float* cn_w   = (const float*)d_in[7];
    const float* cq_w   = (const float*)d_in[8];
    const float* ck_w   = (const float*)d_in[9];
    const float* cv_w   = (const float*)d_in[10];
    const float* co_w   = (const float*)d_in[11];
    const float* c_gate = (const float*)d_in[12];
    const float* fn_w   = (const float*)d_in[13];
    const float* fq_w   = (const float*)d_in[14];
    const float* fk_w   = (const float*)d_in[15];
    const float* fv_w   = (const float*)d_in[16];
    const float* fo_w   = (const float*)d_in[17];
    const float* f_gate = (const float*)d_in[18];
    const int*   dmap   = (const int*)d_in[19];
    float* outp = (float*)d_out;

    char* ws = (char*)d_ws;
    size_t off = 0;
    auto alloc = [&](size_t bytes) -> void* {
        void* p = ws + off;
        off += (bytes + 255) & ~(size_t)255;
        return p;
    };
    __bf16* normed  = (__bf16*)alloc((size_t)CBS * CH * 2);   // reused both stages
    __bf16* qbuf    = (__bf16*)alloc((size_t)CBS * CH * 2);   // reused both stages
    __bf16* kbuf    = (__bf16*)alloc((size_t)CBS * CKV * 2);  // fine-size, reused
    __bf16* vbuf    = (__bf16*)alloc((size_t)CBS * CKV * 2);
    __bf16* attnbuf = (__bf16*)alloc((size_t)CBS * CH * 2);
    float*  outbuf  = (float*)alloc((size_t)CBS * CH * 4);    // stage-1 residual result
    float*  ds_acc  = (float*)alloc((size_t)CB * CND * 4);
    float*  fbias   = (float*)alloc((size_t)CB * CT * 4);

    hipMemsetAsync(ds_acc, 0, (size_t)CB * CND * 4, stream);

    dim3 blk(256);
    dim3 g_big(CH / 128, CBS / 64);          // 16 x 64 (N=2048 GEMMs)
    dim3 g_kvf(CKV / 128, CBS / 64);         // 4 x 64  (N=512 fine K/V)
    dim3 g_kvc(CKV / 128, (CB * CND) / 64);  // 4 x 2   (coarse K/V)
    dim3 g_att(CS / 128, CB * CNH);          // 16 x 32

    // ---- coarse stage ----
    k_rmsnorm<<<CBS, blk, 0, stream>>>(hidden, cn_w, normed);
    k_gemm<__bf16, false><<<g_big, blk, 0, stream>>>(normed, cq_w, nullptr, nullptr,
                                                     nullptr, qbuf, CBS, CH, CH);
    k_gemm<float, false><<<g_kvc, blk, 0, stream>>>(dkey, ck_w, nullptr, nullptr,
                                                    nullptr, kbuf, CB * CND, CKV, CH);
    k_gemm<float, false><<<g_kvc, blk, 0, stream>>>(dval, cv_w, nullptr, nullptr,
                                                    nullptr, vbuf, CB * CND, CKV, CH);
    k_attn_coarse<<<g_att, blk, 0, stream>>>(qbuf, kbuf, vbuf, dmask, ds_acc, attnbuf);
    k_gemm<__bf16, true><<<g_big, blk, 0, stream>>>(attnbuf, co_w, hidden, c_gate,
                                                    outbuf, nullptr, CBS, CH, CH);

    // ---- fine stage ----
    k_fine_bias<<<(CB * CT + 255) / 256, blk, 0, stream>>>(ds_acc, dmap, tmask, fbias);
    k_rmsnorm<<<CBS, blk, 0, stream>>>(outbuf, fn_w, normed);
    k_gemm<__bf16, false><<<g_big, blk, 0, stream>>>(normed, fq_w, nullptr, nullptr,
                                                     nullptr, qbuf, CBS, CH, CH);
    k_gemm<float, false><<<g_kvf, blk, 0, stream>>>(tkey, fk_w, nullptr, nullptr,
                                                    nullptr, kbuf, CB * CT, CKV, CH);
    k_gemm<float, false><<<g_kvf, blk, 0, stream>>>(tval, fv_w, nullptr, nullptr,
                                                    nullptr, vbuf, CB * CT, CKV, CH);
    k_attn_fine<<<g_att, blk, 0, stream>>>(qbuf, kbuf, vbuf, fbias, attnbuf);
    k_gemm<__bf16, true><<<g_big, blk, 0, stream>>>(attnbuf, fo_w, outbuf, f_gate,
                                                    outp, nullptr, CBS, CH, CH);
}
